// Seq2Seq_15779709845796
// MI455X (gfx1250) — matrix operand order repaired and checked by reading, not compiled, not run on hardware
//
#include <hip/hip_runtime.h>
#include <hip/hip_bf16.h>

// ---------------------------------------------------------------------------
// Types for CDNA5 WMMA (wave32)
// ---------------------------------------------------------------------------
typedef __attribute__((ext_vector_type(16))) __bf16 v16bf;
typedef __attribute__((ext_vector_type(8)))  float  v8f;

union FragBF {
    v16bf v;
    uint4 q[2];
    unsigned short u[16];
};

__device__ __forceinline__ unsigned short f2bf(float f) {
    unsigned u = __builtin_bit_cast(unsigned, f);
    unsigned r = (u + 0x7FFFu + ((u >> 16) & 1u)) >> 16;   // round-to-nearest-even
    return (unsigned short)r;
}
__device__ __forceinline__ float bf2f(unsigned short u) {
    unsigned x = ((unsigned)u) << 16;
    return __builtin_bit_cast(float, x);
}
__device__ __forceinline__ float sigmoidf(float x) {
    return 1.0f / (1.0f + __expf(-x));
}

// A fragment: 16x32 bf16 (MxK).  Lane m (0..15) + group g (lane>>4):
//   element e (=2v+h): k = (e<8 ? e : 8+e) + 8*g  -> two contiguous 8-elem runs.
__device__ __forceinline__ v16bf load_A(const unsigned short* __restrict__ base,
                                        int lda, int row0, int k0, int lane) {
    int m = lane & 15, g = lane >> 4;
    const unsigned short* p = base + (size_t)(row0 + m) * lda + k0 + 8 * g;
    FragBF f;
    f.q[0] = *(const uint4*)(p);
    f.q[1] = *(const uint4*)(p + 16);
    return f.v;
}

// B fragment: 32x16 bf16 (KxN) taken as W^T where W is (N x K) row-major:
//   lane column n = lane&15, group g: element e holds k = k0 + (e<8 ? e : 8+e) + 8*g
//   -> two contiguous 8-elem runs of W's row n (same as A).
__device__ __forceinline__ v16bf load_B(const unsigned short* __restrict__ W,
                                        int ldw, int n0, int k0, int lane) {
    int n = lane & 15, g = lane >> 4;
    const unsigned short* p = W + (size_t)(n0 + n) * ldw + k0 + 8 * g;
    FragBF f;
    f.q[0] = *(const uint4*)(p);
    f.q[1] = *(const uint4*)(p + 16);
    return f.v;
}

__device__ __forceinline__ v8f wmma_bf16(v16bf a, v16bf b, v8f c) {
    return __builtin_amdgcn_wmma_f32_16x16x32_bf16(
        /*neg_a=*/false, a, /*neg_b=*/false, b,
        /*c_mod=*/(short)0, c, /*reuse_a=*/false, /*reuse_b=*/false);
}

// ---------------------------------------------------------------------------
// LSTM cell building blocks (one 1024-thread workgroup = 32 waves)
// g = A1 @ W1^T + A2 @ W2^T + bi + bh      -> gbuf (32 x 1024, f32)
// tiles: 2 M-tiles x 64 N-tiles = 128 tiles; 4 tiles per wave.
// NOTE: the tile loop is forced to unroll=1 — with 1024 threads/WG the fully
// unrolled version keeps 4 accumulators + many fragments live and spills to
// scratch (seen as "Folded Reload" scratch_load_b128 in round-1 asm).
// ---------------------------------------------------------------------------
__device__ void cell_gemm(const unsigned short* __restrict__ A1, int lda1, int K1,
                          const unsigned short* __restrict__ W1,
                          const unsigned short* __restrict__ A2, int lda2, int K2,
                          const unsigned short* __restrict__ W2,
                          const float* __restrict__ bi, const float* __restrict__ bh,
                          float* __restrict__ gbuf, int wave, int lane) {
#pragma unroll 1
    for (int j = 0; j < 4; ++j) {
        int tile = wave * 4 + j;
        int mt = tile & 1, nt = tile >> 1;     // nt in 0..63
        int row0 = mt * 16, n0 = nt * 16;
        v8f acc = {};
#pragma unroll
        for (int k0 = 0; k0 < K1; k0 += 32) {
            v16bf a = load_A(A1, lda1, row0, k0, lane);
            v16bf b = load_B(W1, K1, n0, k0, lane);
            acc = wmma_bf16(a, b, acc);
        }
#pragma unroll
        for (int k0 = 0; k0 < K2; k0 += 32) {
            v16bf a = load_A(A2, lda2, row0, k0, lane);
            v16bf b = load_B(W2, K2, n0, k0, lane);
            acc = wmma_bf16(a, b, acc);
        }
        // D layout: lane n = lane&15, g = lane>>4; acc[v] is (M = v+8g, N = n)
        int g = lane >> 4, n = lane & 15;
        float bias = bi[n0 + n] + bh[n0 + n];
#pragma unroll
        for (int v = 0; v < 8; ++v) {
            int brow = row0 + v + 8 * g;
            gbuf[brow * 1024 + n0 + n] = acc[v] + bias;
        }
    }
}

// c = sigmoid(f)*c + sigmoid(i)*tanh(g);  h = sigmoid(o)*tanh(c)
// h stored bf16 (matmul A operand); optionally also f32 (enc_out).
__device__ void lstm_update(const float* __restrict__ gbuf,
                            float* __restrict__ c, unsigned short* __restrict__ hbf,
                            float* __restrict__ hf, int hstride, int tid) {
    for (int idx = tid; idx < 32 * 256; idx += 1024) {
        int b = idx >> 8, hh = idx & 255;
        const float* gr = gbuf + b * 1024;
        float iv = sigmoidf(gr[hh]);
        float fv = sigmoidf(gr[256 + hh]);
        float gv = tanhf(gr[512 + hh]);
        float ov = sigmoidf(gr[768 + hh]);
        float cc = fv * c[idx] + iv * gv;
        c[idx] = cc;
        float hv = ov * tanhf(cc);
        hbf[idx] = f2bf(hv);
        if (hf) hf[(size_t)b * hstride + hh] = hv;
    }
}

// ---------------------------------------------------------------------------
// Prep kernels
// ---------------------------------------------------------------------------
__global__ void cvt_kernel(const float* __restrict__ src,
                           unsigned short* __restrict__ dst, int n) {
    int i = blockIdx.x * blockDim.x + threadIdx.x;
    if (i < n) dst[i] = f2bf(src[i]);
}

__global__ void emb_src_kernel(const int* __restrict__ src,
                               const float* __restrict__ emb,
                               unsigned short* __restrict__ x) {
    int idx = blockIdx.x * blockDim.x + threadIdx.x;   // 32*128*128
    if (idx >= 32 * 128 * 128) return;
    int e = idx & 127;
    int bs = idx >> 7;            // b*S + s
    int tok = src[bs];
    x[idx] = f2bf(emb[(size_t)tok * 128 + e]);
}

__global__ void zero_out_kernel(float* __restrict__ out) {
    int idx = blockIdx.x * blockDim.x + threadIdx.x;   // 32*32000
    if (idx >= 32 * 32000) return;
    int b = idx / 32000, v = idx - b * 32000;
    out[((size_t)b * 64 + 0) * 32000 + v] = 0.f;
}

__global__ void scores_kernel(const float* __restrict__ enc_out,
                              const float* __restrict__ attn_W,
                              float* __restrict__ enc_scores) {
    int idx = blockIdx.x * blockDim.x + threadIdx.x;   // 32*128
    if (idx >= 32 * 128) return;
    const float* eo = enc_out + (size_t)idx * 256;
    float acc = 0.f;
    for (int h = 0; h < 256; ++h) acc += eo[h] * attn_W[h];
    enc_scores[idx] = acc;
}

// ---------------------------------------------------------------------------
// Encoder: single persistent workgroup, 128 time steps of a 2-layer LSTM.
// ---------------------------------------------------------------------------
__global__ __launch_bounds__(1024) void enc_kernel(
    const unsigned short* __restrict__ xsrc,   // (b*S+s)*E bf16
    const unsigned short* __restrict__ wE0i, const unsigned short* __restrict__ wE0h,
    const unsigned short* __restrict__ wE1i, const unsigned short* __restrict__ wE1h,
    const float* __restrict__ b0i, const float* __restrict__ b0h,
    const float* __restrict__ b1i, const float* __restrict__ b1h,
    float* __restrict__ gbuf, float* __restrict__ c0, float* __restrict__ c1,
    unsigned short* __restrict__ h0, unsigned short* __restrict__ h1,
    float* __restrict__ enc_out) {
    int tid = threadIdx.x, lane = tid & 31, wave = tid >> 5;
    for (int i = tid; i < 32 * 256; i += 1024) {
        c0[i] = 0.f; c1[i] = 0.f; h0[i] = 0; h1[i] = 0;
    }
    __threadfence(); __syncthreads();

#pragma unroll 1
    for (int s = 0; s < 128; ++s) {
        // layer 0: x_t (lda = S*E) and h0
        cell_gemm(xsrc + s * 128, 128 * 128, 128, wE0i,
                  h0, 256, 256, wE0h, b0i, b0h, gbuf, wave, lane);
        __threadfence(); __syncthreads();
        lstm_update(gbuf, c0, h0, nullptr, 0, tid);
        __threadfence(); __syncthreads();
        // layer 1: h0 and h1; also emit enc_out[b, s, :] in f32
        cell_gemm(h0, 256, 256, wE1i, h1, 256, 256, wE1h, b1i, b1h, gbuf, wave, lane);
        __threadfence(); __syncthreads();
        lstm_update(gbuf, c1, h1, enc_out + s * 256, 128 * 256, tid);
        __threadfence(); __syncthreads();
    }
}

// ---------------------------------------------------------------------------
// Decoder cell: attention + 2-layer LSTM for one step (single workgroup).
// ---------------------------------------------------------------------------
__global__ __launch_bounds__(1024) void dec_cell_kernel(
    int t, const int* __restrict__ tgt,
    const float* __restrict__ emb_tgt,
    const float* __restrict__ enc_scores, const float* __restrict__ enc_out,
    const float* __restrict__ attn_W, const float* __restrict__ attn_b,
    const unsigned short* __restrict__ wD0i, const unsigned short* __restrict__ wD0h,
    const unsigned short* __restrict__ wD1i, const unsigned short* __restrict__ wD1h,
    const float* __restrict__ b0i, const float* __restrict__ b0h,
    const float* __restrict__ b1i, const float* __restrict__ b1h,
    float* __restrict__ gbuf, float* __restrict__ c0, float* __restrict__ c1,
    unsigned short* __restrict__ h0, unsigned short* __restrict__ h1,
    unsigned short* __restrict__ xin) {
    __shared__ float s_w[32 * 128];   // softmax weights (16 KB)
    __shared__ float s_ah[32];
    int tid = threadIdx.x, lane = tid & 31, wave = tid >> 5;

    // 1) ah[b] = h1[b] . Wa_h + attn_b   (wave w handles batch b = w)
    {
        int b = wave;
        float acc = 0.f;
        for (int h = lane; h < 256; h += 32)
            acc += bf2f(h1[b * 256 + h]) * attn_W[256 + h];
        for (int off = 16; off; off >>= 1) acc += __shfl_xor(acc, off, 32);
        if (lane == 0) s_ah[b] = acc + attn_b[0];
    }
    __syncthreads();

    // 2) softmax over S=128 (4 scores per lane)
    {
        int b = wave;
        float v[4];
        float mx = -3.4e38f;
#pragma unroll
        for (int i = 0; i < 4; ++i) {
            v[i] = enc_scores[b * 128 + lane + i * 32] + s_ah[b];
            mx = fmaxf(mx, v[i]);
        }
        for (int off = 16; off; off >>= 1) mx = fmaxf(mx, __shfl_xor(mx, off, 32));
        float sum = 0.f;
#pragma unroll
        for (int i = 0; i < 4; ++i) { v[i] = __expf(v[i] - mx); sum += v[i]; }
        for (int off = 16; off; off >>= 1) sum += __shfl_xor(sum, off, 32);
        float inv = 1.f / sum;
#pragma unroll
        for (int i = 0; i < 4; ++i) s_w[b * 128 + lane + i * 32] = v[i] * inv;
    }
    __syncthreads();

    // 3) ctx[b,h] = sum_s w[b,s] * enc_out[b,s,h]  -> xin[:,128..383] (bf16)
    for (int idx = tid; idx < 32 * 256; idx += 1024) {
        int b = idx >> 8, h = idx & 255;
        const float* eo = enc_out + (size_t)b * (128 * 256) + h;
        const float* wr = s_w + b * 128;
        float acc = 0.f;
        for (int s2 = 0; s2 < 128; ++s2) acc += wr[s2] * eo[s2 * 256];
        xin[b * 384 + 128 + h] = f2bf(acc);
    }
    // embedding -> xin[:,0..127]
    for (int idx = tid; idx < 32 * 128; idx += 1024) {
        int b = idx >> 7, e = idx & 127;
        int tok = tgt[b * 64 + t];
        xin[b * 384 + e] = f2bf(emb_tgt[(size_t)tok * 128 + e]);
    }
    __threadfence(); __syncthreads();

    // 4) LSTM layer 0 (K = 384 + 256)
    cell_gemm(xin, 384, 384, wD0i, h0, 256, 256, wD0h, b0i, b0h, gbuf, wave, lane);
    __threadfence(); __syncthreads();
    lstm_update(gbuf, c0, h0, nullptr, 0, tid);
    __threadfence(); __syncthreads();

    // 5) LSTM layer 1
    cell_gemm(h0, 256, 256, wD1i, h1, 256, 256, wD1h, b1i, b1h, gbuf, wave, lane);
    __threadfence(); __syncthreads();
    lstm_update(gbuf, c1, h1, nullptr, 0, tid);
}

// ---------------------------------------------------------------------------
// fc projection: pred = h1 @ fc_W^T + fc_b  (M=32, K=256, N=32000)
// 250 workgroups x 8 waves; each wave: one N-tile, both M-tiles, 2x8 WMMAs.
// Writes out[:, t+1, :].
// ---------------------------------------------------------------------------
__global__ __launch_bounds__(256) void fc_kernel(
    const unsigned short* __restrict__ h1, const unsigned short* __restrict__ wFC,
    const float* __restrict__ fc_b, float* __restrict__ out, int t) {
    int lane = threadIdx.x & 31, wave = threadIdx.x >> 5;
    int nt = blockIdx.x * 8 + wave;        // 0..1999
    int n0 = nt * 16;
    v8f acc0 = {}, acc1 = {};
#pragma unroll
    for (int k0 = 0; k0 < 256; k0 += 32) {
        v16bf b  = load_B(wFC, 256, n0, k0, lane);
        v16bf a0 = load_A(h1, 256, 0,  k0, lane);
        v16bf a1 = load_A(h1, 256, 16, k0, lane);
        acc0 = wmma_bf16(a0, b, acc0);
        acc1 = wmma_bf16(a1, b, acc1);
    }
    int g = lane >> 4, n = lane & 15;
    float bias = fc_b[n0 + n];
#pragma unroll
    for (int v = 0; v < 8; ++v) {
        int r0 = v + 8 * g;                // rows 0..15
        out[((size_t)r0 * 64 + (t + 1)) * 32000 + n0 + n]        = acc0[v] + bias;
        out[((size_t)(r0 + 16) * 64 + (t + 1)) * 32000 + n0 + n] = acc1[v] + bias;
    }
}

// ---------------------------------------------------------------------------
// Host launcher
// ---------------------------------------------------------------------------
extern "C" void kernel_launch(void* const* d_in, const int* in_sizes, int n_in,
                              void* d_out, int out_size, void* d_ws, size_t ws_size,
                              hipStream_t stream) {
    const int*   src      = (const int*)d_in[0];
    const int*   tgt      = (const int*)d_in[1];
    const float* emb_src  = (const float*)d_in[2];
    const float* enc_Wih0 = (const float*)d_in[3];
    const float* enc_Whh0 = (const float*)d_in[4];
    const float* enc_bih0 = (const float*)d_in[5];
    const float* enc_bhh0 = (const float*)d_in[6];
    const float* enc_Wih1 = (const float*)d_in[7];
    const float* enc_Whh1 = (const float*)d_in[8];
    const float* enc_bih1 = (const float*)d_in[9];
    const float* enc_bhh1 = (const float*)d_in[10];
    const float* emb_tgt  = (const float*)d_in[11];
    const float* dec_Wih0 = (const float*)d_in[12];
    const float* dec_Whh0 = (const float*)d_in[13];
    const float* dec_bih0 = (const float*)d_in[14];
    const float* dec_bhh0 = (const float*)d_in[15];
    const float* dec_Wih1 = (const float*)d_in[16];
    const float* dec_Whh1 = (const float*)d_in[17];
    const float* dec_bih1 = (const float*)d_in[18];
    const float* dec_bhh1 = (const float*)d_in[19];
    const float* fc_W     = (const float*)d_in[20];
    const float* fc_b     = (const float*)d_in[21];
    const float* attn_W   = (const float*)d_in[22];
    const float* attn_b   = (const float*)d_in[23];
    float* out = (float*)d_out;

    // ---- workspace carve-out ----
    char* ws = (char*)d_ws;
    size_t off = 0;
    auto carve = [&](size_t bytes) -> void* {
        void* p = ws + off;
        off = (off + bytes + 255) & ~(size_t)255;
        return p;
    };
    unsigned short* wE0i = (unsigned short*)carve(1024 * 128 * 2);
    unsigned short* wE0h = (unsigned short*)carve(1024 * 256 * 2);
    unsigned short* wE1i = (unsigned short*)carve(1024 * 256 * 2);
    unsigned short* wE1h = (unsigned short*)carve(1024 * 256 * 2);
    unsigned short* wD0i = (unsigned short*)carve(1024 * 384 * 2);
    unsigned short* wD0h = (unsigned short*)carve(1024 * 256 * 2);
    unsigned short* wD1i = (unsigned short*)carve(1024 * 256 * 2);
    unsigned short* wD1h = (unsigned short*)carve(1024 * 256 * 2);
    unsigned short* wFC  = (unsigned short*)carve((size_t)32000 * 256 * 2);
    unsigned short* xsrc = (unsigned short*)carve(32 * 128 * 128 * 2);
    unsigned short* h0   = (unsigned short*)carve(32 * 256 * 2);
    unsigned short* h1   = (unsigned short*)carve(32 * 256 * 2);
    unsigned short* xin  = (unsigned short*)carve(32 * 384 * 2);
    float* c0      = (float*)carve(32 * 256 * 4);
    float* c1      = (float*)carve(32 * 256 * 4);
    float* gbuf    = (float*)carve(32 * 1024 * 4);
    float* enc_out = (float*)carve((size_t)32 * 128 * 256 * 4);
    float* enc_sc  = (float*)carve(32 * 128 * 4);
    (void)ws_size; (void)n_in; (void)in_sizes; (void)out_size;

    // ---- weight conversion (f32 -> bf16) ----
    auto cvt = [&](const float* s, unsigned short* d, int n) {
        cvt_kernel<<<(n + 255) / 256, 256, 0, stream>>>(s, d, n);
    };
    cvt(enc_Wih0, wE0i, 1024 * 128);
    cvt(enc_Whh0, wE0h, 1024 * 256);
    cvt(enc_Wih1, wE1i, 1024 * 256);
    cvt(enc_Whh1, wE1h, 1024 * 256);
    cvt(dec_Wih0, wD0i, 1024 * 384);
    cvt(dec_Whh0, wD0h, 1024 * 256);
    cvt(dec_Wih1, wD1i, 1024 * 256);
    cvt(dec_Whh1, wD1h, 1024 * 256);
    cvt(fc_W,     wFC,  32000 * 256);

    // source embedding gather -> bf16
    {
        int n = 32 * 128 * 128;
        emb_src_kernel<<<(n + 255) / 256, 256, 0, stream>>>(src, emb_src, xsrc);
    }
    // out[:, 0, :] = 0
    {
        int n = 32 * 32000;
        zero_out_kernel<<<(n + 255) / 256, 256, 0, stream>>>(out);
    }

    // ---- encoder (persistent single workgroup) ----
    enc_kernel<<<1, 1024, 0, stream>>>(xsrc, wE0i, wE0h, wE1i, wE1h,
                                       enc_bih0, enc_bhh0, enc_bih1, enc_bhh1,
                                       gbuf, c0, c1, h0, h1, enc_out);

    // attention score precompute: enc_scores = enc_out . Wa_e
    scores_kernel<<<16, 256, 0, stream>>>(enc_out, attn_W, enc_sc);

    // ---- decoder: 63 steps ----
    for (int t = 0; t < 63; ++t) {
        dec_cell_kernel<<<1, 1024, 0, stream>>>(
            t, tgt, emb_tgt, enc_sc, enc_out, attn_W, attn_b,
            wD0i, wD0h, wD1i, wD1h,
            dec_bih0, dec_bhh0, dec_bih1, dec_bhh1,
            gbuf, c0, c1, h0, h1, xin);
        fc_kernel<<<250, 256, 0, stream>>>(h1, wFC, fc_b, out, t);
    }
}
